// MLP_NW_dist_att_32418413150559
// MI455X (gfx1250) — compile-verified, matrix operand order checked by reading, not run on hardware
//
#include <hip/hip_runtime.h>
#include <hip/hip_fp16.h>

#define BB 8
#define NF 4096
#define NT 4096
#define D_IN 3
#define D_HID 128
#define D_OUT 32
#define D_V 16

#define KI 32          // keys per tile
#define WAVES 4
#define JT (WAVES*16)  // 64 targets per block
#define ROWB 144       // LDS row stride in bytes (36 floats, 16B aligned)

typedef __attribute__((ext_vector_type(16))) _Float16 v16h;
typedef __attribute__((ext_vector_type(8)))  float    v8f;
typedef __attribute__((ext_vector_type(4)))  float    v4f;

// ---------------- Kernel 1: tiny MLP (coords -> 32-dim embedding) ----------------
__global__ __launch_bounds__(256)
void mlp_embed_kernel(const float* __restrict__ coords,  // [BB, N, 3]
                      const float* __restrict__ W1,      // [3, 128]
                      const float* __restrict__ b1,      // [128]
                      const float* __restrict__ W2,      // [128, 32]
                      const float* __restrict__ b2,      // [32]
                      float* __restrict__ out,           // [BB, N, 32]
                      int npts)
{
    int p = blockIdx.x * blockDim.x + threadIdx.x;
    if (p >= npts) return;
    const float c0 = coords[p*3+0], c1 = coords[p*3+1], c2 = coords[p*3+2];
    float acc[D_OUT];
    #pragma unroll
    for (int j = 0; j < D_OUT; ++j) acc[j] = b2[j];
    for (int h = 0; h < D_HID; ++h) {   // weight indices uniform -> scalar loads
        float hv = fmaf(c0, W1[h],
                   fmaf(c1, W1[D_HID + h],
                   fmaf(c2, W1[2*D_HID + h], b1[h])));
        hv = fmaxf(hv, 0.0f);
        #pragma unroll
        for (int j = 0; j < D_OUT; ++j)
            acc[j] = fmaf(hv, W2[h*D_OUT + j], acc[j]);
    }
    #pragma unroll
    for (int j = 0; j < D_OUT; ++j) out[(size_t)p*D_OUT + j] = acc[j];
}

// ---------------- Kernel 2: flash-style L1-distance attention ----------------
__global__ __launch_bounds__(128, 4)
void attn_kernel(const float* __restrict__ wk,    // [BB,NF,32]
                 const float* __restrict__ wq,    // [BB,NT,32]
                 const float* __restrict__ vals,  // [BB,NF,16]
                 float* __restrict__ out)         // [BB,NT,16]
{
    __shared__ float lds[KI * 36];   // 32 rows, 36-float stride
    const int b     = blockIdx.y;
    const int jbase = blockIdx.x * JT;
    const int tid   = threadIdx.x;
    const int lane  = tid & 31;
    const int wave  = tid >> 5;
    const int half  = lane >> 4;     // which 16-lane half
    const int jl    = lane & 15;     // j within the wave's 16-target fragment
    const int jglob = jbase + wave*16 + jl;

    // Pin this lane's Wq row (32 f32) in VGPRs
    v4f wqv[8];
    const float* wqp = wq + ((size_t)b * NT + jglob) * D_OUT;
    #pragma unroll
    for (int q = 0; q < 8; ++q) wqv[q] = ((const v4f*)wqp)[q];

    v8f acc = {0.f,0.f,0.f,0.f,0.f,0.f,0.f,0.f};
    float m = -INFINITY;   // running max (shared between lane halves of same j)
    float l = 0.0f;        // partial exp-sum over THIS lane's key subset

    const float* wkb = wk   + (size_t)b * NF * D_OUT;  // uniform
    const float* vb  = vals + (size_t)b * NF * D_V;    // uniform

    // loop-invariant per-lane 32-bit offsets (uniform base advances with i0)
    const int row     = tid >> 2;
    const int dof     = (tid & 3) * 8;
    const int thr_off = row * D_OUT + dof;             // wk stage offset (floats)
    const int voff    = jl + (half << 3) * D_V;        // values offset (floats)
    const char* kbase0 = (const char*)lds + (half << 3) * ROWB;
    float* ldsT = &lds[row*36 + dof];

    #pragma clang loop unroll(disable)
    for (int i0 = 0; i0 < NF; i0 += KI) {
        // cooperative stage of Wk tile [32 keys][32 dims] into LDS
        {
            const float* srcu = wkb + (size_t)i0 * D_OUT;   // uniform (SGPR base)
            v4f a0 = *(const v4f*)(srcu + thr_off);
            v4f a1 = *(const v4f*)(srcu + thr_off + 4);
            *(v4f*)ldsT       = a0;
            *(v4f*)(ldsT + 4) = a1;
        }
        __syncthreads();

        // L1 distances: d-chunk outer, 16 slots inner; sched_barrier bounds the
        // scheduler to one 16-load group in flight (keeps VGPR demand low)
        float d[16];
        #pragma unroll
        for (int p = 0; p < 16; ++p) d[p] = 0.0f;
        #pragma unroll
        for (int q = 0; q < 8; ++q) {
            const v4f wv = wqv[q];
            #pragma unroll
            for (int p = 0; p < 16; ++p) {
                const v4f kv = *(const v4f*)(kbase0 + (p + (p & 8)) * ROWB + q * 16);
                d[p] += fabsf(kv.x - wv.x) + fabsf(kv.y - wv.y)
                      + fabsf(kv.z - wv.z) + fabsf(kv.w - wv.w);
            }
            __builtin_amdgcn_sched_barrier(0);
        }

        // B fragment (values tile): uniform base + invariant lane offset + imm
        float bfl[16];
        {
            const float* vsrcu = vb + (size_t)i0 * D_V;     // uniform (SGPR base)
            #pragma unroll
            for (int p = 0; p < 16; ++p)
                bfl[p] = vsrcu[voff + (p + (p & 8)) * D_V];
        }

        // scores in place: d[p] <- -(d^2)/2
        float tmax = -INFINITY;
        #pragma unroll
        for (int p = 0; p < 16; ++p) {
            float sc = -0.5f * d[p] * d[p];
            d[p] = sc;
            tmax = fmaxf(tmax, sc);
        }
        // share tile max across lane halves (same j in lane and lane^16)
        tmax = fmaxf(tmax, __shfl_xor(tmax, 16, 32));
        float mnew  = fmaxf(m, tmax);
        float scale = __expf(m - mnew);   // exp(-inf)=0 on first tile
        m = mnew;

        // A fragment: P = exp(s - m) in f16; 4-way partial sums shorten the
        // serial dependence chain of the row-sum accumulation
        v16h a;
        float l0 = 0.f, l1 = 0.f, l2 = 0.f, l3 = 0.f;
        #pragma unroll
        for (int p = 0; p < 16; p += 4) {
            float e0 = __expf(d[p+0] - mnew);
            float e1 = __expf(d[p+1] - mnew);
            float e2 = __expf(d[p+2] - mnew);
            float e3 = __expf(d[p+3] - mnew);
            l0 += e0; l1 += e1; l2 += e2; l3 += e3;
            a[p+0] = (_Float16)e0;
            a[p+1] = (_Float16)e1;
            a[p+2] = (_Float16)e2;
            a[p+3] = (_Float16)e3;
        }
        l = fmaf(l, scale, (l0 + l1) + (l2 + l3));

        // B fragment to f16
        v16h bf;
        #pragma unroll
        for (int p = 0; p < 16; ++p)
            bf[p] = (_Float16)bfl[p];

        // rescale accumulator: D row r holds j = r + 8*half; scale[j] lives at lane j
        #pragma unroll
        for (int r = 0; r < 8; ++r) {
            float srow = __shfl(scale, (half << 3) + r, 32);
            acc[r] *= srow;
        }

        acc = __builtin_amdgcn_wmma_f32_16x16x32_f16(
            false, a, false, bf, (short)0, acc, false, false);

        __syncthreads();
    }

    // combine the two half-lane partial sums, normalize, store
    float ltot = l + __shfl_xor(l, 16, 32);
    float inv  = __builtin_amdgcn_rcpf(ltot);   // ~1 ulp, << f16-P quantization
    // base pointer folds row = 8*half; per-r rows become immediate offsets r*16
    float* ob = out + ((size_t)b * NT + jbase + wave*16 + (half << 3)) * D_V + jl;
    #pragma unroll
    for (int r = 0; r < 8; ++r) {
        float invr = __shfl(inv, (half << 3) + r, 32);
        ob[r * D_V] = acc[r] * invr;
    }
}

// ---------------- Launcher ----------------
extern "C" void kernel_launch(void* const* d_in, const int* in_sizes, int n_in,
                              void* d_out, int out_size, void* d_ws, size_t ws_size,
                              hipStream_t stream) {
    const float* coords_f = (const float*)d_in[0];   // [BB,NF,3]
    const float* values_f = (const float*)d_in[1];   // [BB,NF,16]
    const float* coords_t = (const float*)d_in[2];   // [BB,NT,3]
    const float* Wk1 = (const float*)d_in[3];
    const float* bk1 = (const float*)d_in[4];
    const float* Wk2 = (const float*)d_in[5];
    const float* bk2 = (const float*)d_in[6];
    const float* Wq1 = (const float*)d_in[7];
    const float* bq1 = (const float*)d_in[8];
    const float* Wq2 = (const float*)d_in[9];
    const float* bq2 = (const float*)d_in[10];

    float* wkbuf = (float*)d_ws;                     // [BB,NF,32] = 4 MB
    float* wqbuf = wkbuf + (size_t)BB * NF * D_OUT;  // [BB,NT,32] = 4 MB

    int nk = BB * NF;
    int nq = BB * NT;
    mlp_embed_kernel<<<(nk + 255) / 256, 256, 0, stream>>>(
        coords_f, Wk1, bk1, Wk2, bk2, wkbuf, nk);
    mlp_embed_kernel<<<(nq + 255) / 256, 256, 0, stream>>>(
        coords_t, Wq1, bq1, Wq2, bq2, wqbuf, nq);

    dim3 grid(NT / JT, BB);
    attn_kernel<<<grid, 128, 0, stream>>>(wkbuf, wqbuf, values_f, (float*)d_out);
}
